// GlobalMapEncoder_33595234190024
// MI455X (gfx1250) — compile-verified, matrix operand order checked by reading, not run on hardware
//
#include <hip/hip_runtime.h>
#include <hip/hip_bf16.h>
#include <cstdint>

// GlobalMapEncoder forward for MI455X (gfx1250, wave32, WMMA).
// End-to-end bf16 dataflow: weights pre-transposed+converted once, all GEMM
// tiles staged bf16 via async global->LDS (ASYNCcnt), v_wmma_f32_16x16x32_bf16
// everywhere, fp32 VALU only for softmax/LayerNorm/epilogues.

#define DEV __device__ __forceinline__

typedef __attribute__((ext_vector_type(16))) __bf16 v16bf;
typedef __attribute__((ext_vector_type(8)))  float  v8f;

namespace gme {

constexpr int B = 32, T = 30, L = 16, N = 128, D = 768, Lt = 80;
constexpr int H = 12, DH = 64, FF = 3072;
constexpr float NEGB = -10000.0f;
constexpr float LNEPS = 1e-12f;

DEV __bf16 f2bf(float f) {
  unsigned u = __float_as_uint(f);
  u += 0x7fffu + ((u >> 16) & 1u);           // round-to-nearest-even
  unsigned short h = (unsigned short)(u >> 16);
  __bf16 r;
  __builtin_memcpy(&r, &h, 2);
  return r;
}

// 16-bit A-matrix fragment K index for dword j (ISA 7.12.2):
// lanes 0-15: K = {0,2,4,6, 16,18,20,22} pairs ; lanes 16-31: +8 / +24
DEV int frag_k(int lane, int j) {
  int kb = (lane & 16) ? 8 : 0;
  return (j < 4) ? (kb + 2 * j) : (kb + 16 + 2 * (j - 4));
}

DEV v8f wmma_bf16(v16bf a, v16bf b, v8f c) {
  return __builtin_amdgcn_wmma_f32_16x16x32_bf16(false, a, false, b, (short)0, c,
                                                 false, false);
}

DEV v8f zero8() { v8f z = {0.f, 0.f, 0.f, 0.f, 0.f, 0.f, 0.f, 0.f}; return z; }

// Async global->LDS copy, 16B per lane (GLOBAL_LOAD_ASYNC_TO_LDS_B128, GV mode).
DEV void async_load_b128(unsigned lds_off, const void* gaddr) {
  asm volatile("global_load_async_to_lds_b128 %0, %1, off"
               :: "v"(lds_off), "v"(gaddr)
               : "memory");
}
DEV void wait_async0() { asm volatile("s_wait_asynccnt 0" ::: "memory"); }

// ---------------------------------------------------------------------------
// one-time converters: fp32 -> bf16 (elementwise / transposing for weights)
// ---------------------------------------------------------------------------
__global__ __launch_bounds__(256) void cvt_kernel(
    const float* __restrict__ in, __bf16* __restrict__ out, int n) {
  int i = blockIdx.x * 256 + threadIdx.x;
  if (i < n) out[i] = f2bf(in[i]);
}
// out[n*K + k] = in[k*Ncol + n]  (weight W[K,Ncol] -> W^T[Ncol,K], bf16)
__global__ __launch_bounds__(256) void cvtT_kernel(
    const float* __restrict__ in, __bf16* __restrict__ out, int K, int Ncol) {
  int i = blockIdx.x * 256 + threadIdx.x;
  if (i >= K * Ncol) return;
  int n = i / K, k = i - n * K;
  out[i] = f2bf(in[(size_t)k * Ncol + n]);
}

// ---------------------------------------------------------------------------
// aggregate: segment means of candidate embeddings + visited-node override
// ---------------------------------------------------------------------------
__global__ __launch_bounds__(256) void aggregate_kernel(
    const float* __restrict__ traj, const int* __restrict__ lens,
    const int* __restrict__ cand, float* __restrict__ x) {
  __shared__ int sc[T * L];
  __shared__ int sl[T];
  const int b = blockIdx.x >> 7, n = blockIdx.x & 127;
  const int tid = threadIdx.x;
  for (int i = tid; i < T * L; i += 256) sc[i] = cand[b * T * L + i];
  if (tid < T) sl[tid] = lens[b * T + tid];
  __syncthreads();
  float* out = x + ((size_t)(b * N + n)) * D;
  if (n == 0) {                               // stop-token slot = zeros
    for (int d = tid; d < D; d += 256) out[d] = 0.0f;
    return;
  }
  if (n <= T) {                               // visited node: masked mean of step n-1
    int t = n - 1, len = sl[t];
    float inv = 1.0f / (float)len;
    for (int d = tid; d < D; d += 256) {
      float a = 0.f;
      for (int l = 0; l < len; ++l)
        a += traj[(((size_t)(b * T + t)) * L + l) * D + d];
      out[d] = a * inv;
    }
    return;
  }
  float a0 = 0.f, a1 = 0.f, a2 = 0.f;
  int cnt = 0;
  for (int i = 0; i < T * L; ++i) {
    int t = i >> 4, l = i & 15;
    if (l < sl[t] && sc[i] == n) {
      ++cnt;
      const float* p = traj + ((size_t)(b * T * L + i)) * D;
      a0 += p[tid]; a1 += p[tid + 256]; a2 += p[tid + 512];
    }
  }
  float inv = 1.0f / fmaxf((float)cnt, 1.0f);
  out[tid] = a0 * inv; out[tid + 256] = a1 * inv; out[tid + 512] = a2 * inv;
}

// ---------------------------------------------------------------------------
// x = gmap_img + step_emb[ids] + LayerNorm(pos_fts @ Wp + bp); mirrors to bf16
// ---------------------------------------------------------------------------
__global__ __launch_bounds__(256) void embed_kernel(
    float* __restrict__ x, __bf16* __restrict__ xb, const int* __restrict__ sids,
    const float* __restrict__ semb, const float* __restrict__ fts,
    const float* __restrict__ pw, const float* __restrict__ pb,
    const float* __restrict__ pg, const float* __restrict__ pbeta) {
  __shared__ float red[256];
  const int row = blockIdx.x;
  const int tid = threadIdx.x;
  const float* f = fts + (size_t)row * 7;
  float fl[7];
#pragma unroll
  for (int j = 0; j < 7; ++j) fl[j] = f[j];
  float p[3];
#pragma unroll
  for (int j = 0; j < 3; ++j) {
    int d = tid + j * 256;
    float v = pb[d];
#pragma unroll
    for (int q = 0; q < 7; ++q) v += fl[q] * pw[q * D + d];
    p[j] = v;
  }
  red[tid] = p[0] + p[1] + p[2];
  __syncthreads();
  for (int s = 128; s > 0; s >>= 1) { if (tid < s) red[tid] += red[tid + s]; __syncthreads(); }
  float mu = red[0] / (float)D;
  __syncthreads();
  float d0 = p[0] - mu, d1 = p[1] - mu, d2 = p[2] - mu;
  red[tid] = d0 * d0 + d1 * d1 + d2 * d2;
  __syncthreads();
  for (int s = 128; s > 0; s >>= 1) { if (tid < s) red[tid] += red[tid + s]; __syncthreads(); }
  float rinv = rsqrtf(red[0] / (float)D + LNEPS);
  int sid = sids[row];
  float* xr = x + (size_t)row * D;
  __bf16* xbr = xb + (size_t)row * D;
  const float* se = semb + (size_t)sid * D;
  float dv[3] = {d0, d1, d2};
#pragma unroll
  for (int j = 0; j < 3; ++j) {
    int d = tid + j * 256;
    float v = xr[d] + se[d] + dv[j] * rinv * pg[d] + pbeta[d];
    xr[d] = v;
    xbr[d] = f2bf(v);
  }
}

// ---------------------------------------------------------------------------
// out = LayerNorm(x + delta); optional bf16 mirror; in-place safe
// ---------------------------------------------------------------------------
__global__ __launch_bounds__(256) void add_ln_kernel(
    const float* __restrict__ x, const float* __restrict__ dlt,
    const float* __restrict__ g, const float* __restrict__ beta,
    float* __restrict__ out, __bf16* __restrict__ ob) {
  __shared__ float red[256];
  const int row = blockIdx.x;
  const int tid = threadIdx.x;
  const float* xr = x + (size_t)row * D;
  const float* dr = dlt + (size_t)row * D;
  float v[3];
#pragma unroll
  for (int j = 0; j < 3; ++j) { int d = tid + j * 256; v[j] = xr[d] + dr[d]; }
  red[tid] = v[0] + v[1] + v[2];
  __syncthreads();
  for (int s = 128; s > 0; s >>= 1) { if (tid < s) red[tid] += red[tid + s]; __syncthreads(); }
  float mu = red[0] / (float)D;
  __syncthreads();
  float d0 = v[0] - mu, d1 = v[1] - mu, d2 = v[2] - mu;
  red[tid] = d0 * d0 + d1 * d1 + d2 * d2;
  __syncthreads();
  for (int s = 128; s > 0; s >>= 1) { if (tid < s) red[tid] += red[tid + s]; __syncthreads(); }
  float rinv = rsqrtf(red[0] / (float)D + LNEPS);
  float dv[3] = {d0, d1, d2};
  float* o = out + (size_t)row * D;
#pragma unroll
  for (int j = 0; j < 3; ++j) {
    int d = tid + j * 256;
    float r = dv[j] * rinv * g[d] + beta[d];
    o[d] = r;
    if (ob) ob[(size_t)row * D + d] = f2bf(r);
  }
}

// ---------------------------------------------------------------------------
// pure-bf16 WMMA GEMM: Y[M,Ncol] = epi(A[M,K] @ W[K,Ncol] + bias)
// A: bf16 [M,K] row-major; Wt: bf16 [Ncol,K] (pre-transposed).
// block = 256 threads (8 waves), tile 64(M) x 128(N), K-step 32.
// Tiles staged via async global->LDS; fragments are raw ds loads (no cvt).
// EPI: 0 = fp32 out + bias, 1 = bf16 out + bias, 2 = bf16 out + bias + GELU
// ---------------------------------------------------------------------------
template <int EPI>
__global__ __launch_bounds__(256) void gemm_kernel(
    const __bf16* __restrict__ A, const __bf16* __restrict__ Wt,
    const float* __restrict__ bias, float* __restrict__ Yf,
    __bf16* __restrict__ Yb, int M, int K, int Ncol) {
  // rows padded to 40 halfwords (80B = 16B-aligned for async b128 scatter,
  // 20-bank row stride -> conflict-free fragment reads)
  __shared__ __align__(16) __bf16 sA[64][40];
  __shared__ __align__(16) __bf16 sB[128][40];
  const int tid = threadIdx.x;
  const int lane = tid & 31, wave = tid >> 5;
  const int rm = (wave & 3) * 16, cn = (wave >> 2) * 64;
  const int n0 = blockIdx.x * 128, m0 = blockIdx.y * 64;
  const int mA = lane & 15;
  v8f acc[4];
#pragma unroll
  for (int i = 0; i < 4; ++i) acc[i] = zero8();

  for (int k0 = 0; k0 < K; k0 += 32) {
    // A tile 64x32 bf16 = 256 x 16B chunks, 1 per thread
    {
      int r = tid >> 2, q = tid & 3;
      async_load_b128((unsigned)(uintptr_t)&sA[r][q * 8],
                      &A[(size_t)(m0 + r) * K + k0 + q * 8]);
    }
    // B tile 128x32 bf16 = 512 x 16B chunks, 2 per thread
#pragma unroll
    for (int j = 0; j < 2; ++j) {
      int c = tid + j * 256;
      int n = c >> 2, q = c & 3;
      async_load_b128((unsigned)(uintptr_t)&sB[n][q * 8],
                      &Wt[(size_t)(n0 + n) * K + k0 + q * 8]);
    }
    wait_async0();
    __syncthreads();

    v16bf a;
#pragma unroll
    for (int j = 0; j < 8; ++j) {
      int k = frag_k(lane, j);
      a[2 * j]     = sA[rm + mA][k];
      a[2 * j + 1] = sA[rm + mA][k + 1];
    }
#pragma unroll
    for (int nt = 0; nt < 4; ++nt) {
      v16bf bm;
      int n = cn + nt * 16 + (lane & 15);
#pragma unroll
      for (int j = 0; j < 8; ++j) {
        int k = frag_k(lane, j);
        bm[2 * j]     = sB[n][k];
        bm[2 * j + 1] = sB[n][k + 1];
      }
      acc[nt] = wmma_bf16(a, bm, acc[nt]);
    }
    __syncthreads();
  }
  const int mrow = rm + ((lane & 16) ? 8 : 0);
#pragma unroll
  for (int nt = 0; nt < 4; ++nt) {
    int col = n0 + cn + nt * 16 + (lane & 15);
    float bb = bias[col];
#pragma unroll
    for (int r = 0; r < 8; ++r) {
      float v = acc[nt][r] + bb;
      if (EPI == 2) {
        float x3 = v * v * v;
        v = 0.5f * v * (1.0f + tanhf(0.7978845608028654f * (v + 0.044715f * x3)));
      }
      size_t idx = (size_t)(m0 + mrow + r) * Ncol + col;
      if (EPI == 0) Yf[idx] = v;
      else          Yb[idx] = f2bf(v);
    }
  }
}

// ---------------------------------------------------------------------------
// fused attention per (b,h), bf16 Q/K/V in, bf16 ctx out:
// WMMA QK^T -> bias -> softmax (register, half-wave shfl) ->
// probs via LDS (C-layout -> A-layout) -> WMMA P@V
// grid = B*H blocks, 256 threads (8 waves x 16 query rows)
// ---------------------------------------------------------------------------
template <bool CROSS>
__global__ __launch_bounds__(256) void attn_kernel(
    const __bf16* __restrict__ Q, const __bf16* __restrict__ Kb,
    const __bf16* __restrict__ Vb, __bf16* __restrict__ Ctx,
    const unsigned char* __restrict__ tmask, const int* __restrict__ glens,
    const float* __restrict__ sprels, const float* __restrict__ swp,
    const float* __restrict__ sbp) {
  constexpr int NK  = CROSS ? Lt : N;     // real keys (80 / 128)
  constexpr int NKP = CROSS ? 96 : 128;   // padded to multiple of 32 for P@V
  constexpr int KT  = NK / 16;            // logit tiles (5 / 8)
  constexpr int S2  = NKP / 32;           // P@V K-steps (3 / 4)
  __shared__ __bf16 lP[N * NKP];

  const int b = blockIdx.x / H, h = blockIdx.x % H;
  const int tid = threadIdx.x, lane = tid & 31, wave = tid >> 5;
  const int qbase = wave * 16;
  const int mA = lane & 15;
  const int hi = (lane & 16) ? 8 : 0;

  // Q fragments (A layout), dh = s*32 + k
  const __bf16* qrow = Q + ((size_t)(b * N + qbase + mA)) * D + h * DH;
  v16bf aq[2];
#pragma unroll
  for (int s = 0; s < 2; ++s)
#pragma unroll
    for (int j = 0; j < 8; ++j) {
      int k = frag_k(lane, j);
      aq[s][2 * j]     = qrow[s * 32 + k];
      aq[s][2 * j + 1] = qrow[s * 32 + k + 1];
    }

  float sw = 0.f, sb = 0.f;
  int gl = 0;
  if (!CROSS) { sw = swp[0]; sb = sbp[0]; gl = glens[b]; }

  float logit[KT][8];
#pragma unroll
  for (int kt = 0; kt < KT; ++kt) {
    v8f acc = zero8();
    const int keyB = kt * 16 + (lane & 15);
    const __bf16* krow = Kb + ((size_t)(b * NK + keyB)) * D + h * DH;
#pragma unroll
    for (int s = 0; s < 2; ++s) {
      v16bf bk;
#pragma unroll
      for (int j = 0; j < 8; ++j) {
        int k = frag_k(lane, j);
        bk[2 * j]     = krow[s * 32 + k];
        bk[2 * j + 1] = krow[s * 32 + k + 1];
      }
      acc = wmma_bf16(aq[s], bk, acc);
    }
    const int key = kt * 16 + (lane & 15);   // C-layout column
    float biasv;
    if (CROSS) biasv = tmask[b * Lt + key] ? 0.0f : NEGB;
    else       biasv = (key < gl ? 0.0f : NEGB) + sb;
#pragma unroll
    for (int r = 0; r < 8; ++r) {
      int qg = qbase + r + hi;
      float extra = 0.f;
      if (!CROSS) extra = sw * sprels[((size_t)(b * N) + qg) * N + key];
      logit[kt][r] = acc[r] * 0.125f + biasv + extra;   // 1/sqrt(64)
    }
  }

  // softmax per query row: row lives across the 16 lanes of a half-wave
#pragma unroll
  for (int r = 0; r < 8; ++r) {
    float m = -3.0e38f;
#pragma unroll
    for (int kt = 0; kt < KT; ++kt) m = fmaxf(m, logit[kt][r]);
#pragma unroll
    for (int msk = 1; msk < 16; msk <<= 1) m = fmaxf(m, __shfl_xor(m, msk, 32));
    float s = 0.f;
#pragma unroll
    for (int kt = 0; kt < KT; ++kt) { logit[kt][r] = __expf(logit[kt][r] - m); s += logit[kt][r]; }
#pragma unroll
    for (int msk = 1; msk < 16; msk <<= 1) s += __shfl_xor(s, msk, 32);
    float inv = 1.0f / s;
#pragma unroll
    for (int kt = 0; kt < KT; ++kt) logit[kt][r] *= inv;
  }

  // scatter probs to LDS row-major (zero padding for cross keys 80..95)
  for (int i = tid; i < N * NKP; i += 256) lP[i] = f2bf(0.0f);
  __syncthreads();
#pragma unroll
  for (int kt = 0; kt < KT; ++kt) {
    int key = kt * 16 + (lane & 15);
#pragma unroll
    for (int r = 0; r < 8; ++r) lP[(qbase + r + hi) * NKP + key] = f2bf(logit[kt][r]);
  }
  __syncthreads();

  // ctx = P @ V  (contraction over keys; padded rows multiply by zero probs)
  v8f acc2[4];
#pragma unroll
  for (int i = 0; i < 4; ++i) acc2[i] = zero8();
#pragma unroll
  for (int s2 = 0; s2 < S2; ++s2) {
    v16bf ap;
#pragma unroll
    for (int j = 0; j < 8; ++j) {
      int k = frag_k(lane, j);
      ap[2 * j]     = lP[(qbase + mA) * NKP + s2 * 32 + k];
      ap[2 * j + 1] = lP[(qbase + mA) * NKP + s2 * 32 + k + 1];
    }
#pragma unroll
    for (int nt = 0; nt < 4; ++nt) {
      int dhi = nt * 16 + (lane & 15);
      v16bf bv;
#pragma unroll
      for (int j = 0; j < 8; ++j) {
        int k = frag_k(lane, j);
        int key = s2 * 32 + k;
        bv[2 * j]     = Vb[((size_t)(b * NK) + key) * D + h * DH + dhi];
        bv[2 * j + 1] = Vb[((size_t)(b * NK) + key + 1) * D + h * DH + dhi];
      }
      acc2[nt] = wmma_bf16(ap, bv, acc2[nt]);
    }
  }
#pragma unroll
  for (int nt = 0; nt < 4; ++nt) {
    int dhi = nt * 16 + (lane & 15);
#pragma unroll
    for (int r = 0; r < 8; ++r)
      Ctx[((size_t)(b * N) + qbase + r + hi) * D + h * DH + dhi] = f2bf(acc2[nt][r]);
  }
}

template <int EPI>
static void launch_gemm(const __bf16* A, const __bf16* Wt, const float* bias,
                        float* Yf, __bf16* Yb, int M, int K, int Ncol,
                        hipStream_t stream) {
  dim3 g(Ncol / 128, M / 64), blk(256);
  gemm_kernel<EPI><<<g, blk, 0, stream>>>(A, Wt, bias, Yf, Yb, M, K, Ncol);
}

}  // namespace gme

// ---------------------------------------------------------------------------
// Input index map (setup_inputs order; params flattened as jax tree_leaves,
// i.e. alphabetical dict keys, lists in order):
//   0 txt_embeds  1 txt_masks  2 traj_embeds  3 traj_vp_lens  4 cand_idx
//   5 gmap_step_ids  6 gmap_pos_fts  7 gmap_lens  8 graph_sprels
//   9.. params: 4 layers x 26 leaves
//     [+0]cross.beta [+1]cross.g [+2]cross.k.b [+3]cross.k.w [+4]cross.o.b
//     [+5]cross.o.w  [+6]cross.q.b [+7]cross.q.w [+8]cross.v.b [+9]cross.v.w
//     [+10]ffn_beta [+11]ffn_g [+12]inter.b [+13]inter.w [+14]out.b [+15]out.w
//     [+16..25] self.{beta,g,k.b,k.w,o.b,o.w,q.b,q.w,v.b,v.w}
//   113 pos.b 114 pos.w 115 pos_beta 116 pos_g 117 sprel_b 118 sprel_w 119 step_emb
// ---------------------------------------------------------------------------
extern "C" void kernel_launch(void* const* d_in, const int* in_sizes, int n_in,
                              void* d_out, int out_size, void* d_ws, size_t ws_size,
                              hipStream_t stream) {
  using namespace gme;
  (void)in_sizes; (void)n_in; (void)out_size; (void)ws_size;

  const float* txt            = (const float*)d_in[0];
  const unsigned char* tmask  = (const unsigned char*)d_in[1];
  const float* traj           = (const float*)d_in[2];
  const int*   vplens         = (const int*)d_in[3];
  const int*   cand           = (const int*)d_in[4];
  const int*   sids           = (const int*)d_in[5];
  const float* posfts         = (const float*)d_in[6];
  const int*   glens          = (const int*)d_in[7];
  const float* sprels         = (const float*)d_in[8];
  auto P = [&](int i) { return (const float*)d_in[i]; };

  const size_t RD  = (size_t)B * N * D;    // 3,145,728
  const size_t TXT = (size_t)B * Lt * D;   // 1,966,080
  const size_t DD  = (size_t)D * D;        // 589,824
  const size_t DFF = (size_t)D * FF;       // 2,359,296
  const size_t WLAYER = 8 * DD + 2 * DFF;  // 9,437,184 bf16/layer

  // workspace layout: [fp32: x, delta][bf16: x,q,k,v,c,h, txt, weights]
  float* xbuf = (float*)d_ws;
  float* dbuf = xbuf + RD;
  __bf16* xbf   = (__bf16*)(dbuf + RD);
  __bf16* qbf   = xbf + RD;
  __bf16* kbf   = qbf + RD;
  __bf16* vbf   = kbf + RD;
  __bf16* cbf   = vbf + RD;
  __bf16* hbf   = cbf + RD;        // RD*4 elems
  __bf16* txtbf = hbf + RD * 4;
  __bf16* wbf   = txtbf + TXT;     // 4 * WLAYER elems
  float* obuf = (float*)d_out;

  dim3 blk(256);

  // one-time conversions: txt -> bf16; weights -> bf16 transposed [Ncol][K]
  cvt_kernel<<<(int)((TXT + 255) / 256), blk, 0, stream>>>(txt, txtbf, (int)TXT);
  auto cvtT = [&](const float* w, __bf16* dst, int K, int Ncol) {
    int tot = K * Ncol;
    cvtT_kernel<<<(tot + 255) / 256, blk, 0, stream>>>(w, dst, K, Ncol);
  };
  for (int lp = 0; lp < 4; ++lp) {
    const int pb = 9 + lp * 26;
    __bf16* wl = wbf + (size_t)lp * WLAYER;
    cvtT(P(pb + 7),  wl + 0 * DD, D, D);            // cross q
    cvtT(P(pb + 3),  wl + 1 * DD, D, D);            // cross k
    cvtT(P(pb + 9),  wl + 2 * DD, D, D);            // cross v
    cvtT(P(pb + 5),  wl + 3 * DD, D, D);            // cross o
    cvtT(P(pb + 23), wl + 4 * DD, D, D);            // self q
    cvtT(P(pb + 19), wl + 5 * DD, D, D);            // self k
    cvtT(P(pb + 25), wl + 6 * DD, D, D);            // self v
    cvtT(P(pb + 21), wl + 7 * DD, D, D);            // self o
    cvtT(P(pb + 13), wl + 8 * DD, D, FF);           // inter
    cvtT(P(pb + 15), wl + 8 * DD + DFF, FF, D);     // out
  }

  aggregate_kernel<<<B * N, blk, 0, stream>>>(traj, vplens, cand, xbuf);
  embed_kernel<<<B * N, blk, 0, stream>>>(xbuf, xbf, sids, P(119), posfts,
                                          P(114), P(113), P(116), P(115));

  for (int lp = 0; lp < 4; ++lp) {
    const int pb = 9 + lp * 26;
    __bf16* wl = wbf + (size_t)lp * WLAYER;
    // ---- cross attention: gmap tokens attend to text ----
    launch_gemm<1>(xbf,   wl + 0 * DD, P(pb + 6), nullptr, qbf, B * N,  D, D, stream);
    launch_gemm<1>(txtbf, wl + 1 * DD, P(pb + 2), nullptr, kbf, B * Lt, D, D, stream);
    launch_gemm<1>(txtbf, wl + 2 * DD, P(pb + 8), nullptr, vbf, B * Lt, D, D, stream);
    attn_kernel<true><<<B * H, blk, 0, stream>>>(qbf, kbf, vbf, cbf,
                                                 tmask, glens, sprels, P(118), P(117));
    launch_gemm<0>(cbf, wl + 3 * DD, P(pb + 4), dbuf, nullptr, B * N, D, D, stream);
    add_ln_kernel<<<B * N, blk, 0, stream>>>(xbuf, dbuf, P(pb + 1), P(pb + 0), xbuf, xbf);
    // ---- graph-aware self attention ----
    launch_gemm<1>(xbf, wl + 4 * DD, P(pb + 22), nullptr, qbf, B * N, D, D, stream);
    launch_gemm<1>(xbf, wl + 5 * DD, P(pb + 18), nullptr, kbf, B * N, D, D, stream);
    launch_gemm<1>(xbf, wl + 6 * DD, P(pb + 24), nullptr, vbf, B * N, D, D, stream);
    attn_kernel<false><<<B * H, blk, 0, stream>>>(qbf, kbf, vbf, cbf,
                                                  tmask, glens, sprels, P(118), P(117));
    launch_gemm<0>(cbf, wl + 7 * DD, P(pb + 20), dbuf, nullptr, B * N, D, D, stream);
    add_ln_kernel<<<B * N, blk, 0, stream>>>(xbuf, dbuf, P(pb + 17), P(pb + 16), xbuf, xbf);
    // ---- FFN ----
    launch_gemm<2>(xbf, wl + 8 * DD, P(pb + 12), nullptr, hbf, B * N, D, FF, stream);
    launch_gemm<0>(hbf, wl + 8 * DD + DFF, P(pb + 14), dbuf, nullptr, B * N, FF, D, stream);
    float* tgt = (lp == 3) ? obuf : xbuf;
    __bf16* tgtb = (lp == 3) ? nullptr : xbf;
    add_ln_kernel<<<B * N, blk, 0, stream>>>(xbuf, dbuf, P(pb + 11), P(pb + 10), tgt, tgtb);
  }
}
